// MyRnn_65154653880459
// MI455X (gfx1250) — matrix operand order repaired and checked by reading, not compiled, not run on hardware
//
#include <hip/hip_runtime.h>
#include <hip/hip_bf16.h>
#include <stdint.h>

// Problem sizes (fixed by the reference)
#define B_ 128
#define T_ 1024
#define D_ 512
#define H_ 512

#define STRIDE 520   // 512 + 8 bf16 pad -> 1040B row stride, conflict-free ds_load_b128

typedef __attribute__((ext_vector_type(16))) __bf16 v16bf;
typedef __attribute__((ext_vector_type(8)))  float  v8f;
typedef __attribute__((ext_vector_type(4)))  unsigned int v4u;
typedef __attribute__((ext_vector_type(8)))  int    v8i_t;
typedef __attribute__((ext_vector_type(4)))  int    v4i_t;

#if defined(__HIP_DEVICE_COMPILE__) && __has_builtin(__builtin_amdgcn_tensor_load_to_lds) && __has_builtin(__builtin_amdgcn_s_wait_tensorcnt)
#define USE_TDM 1
#else
#define USE_TDM 0
#endif

// ---------------------------------------------------------------------------
// Kernel A: one-time prep — W1T/W2T = transpose + bf16 convert ([n][k] layout
// so B-fragment loads (lane=n, contiguous k) are 2x16B contiguous reads).
// ---------------------------------------------------------------------------
__global__ void prep_kernel(const float* __restrict__ W1, const float* __restrict__ W2,
                            __bf16* __restrict__ W1T, __bf16* __restrict__ W2T) {
    int idx = blockIdx.x * blockDim.x + threadIdx.x;   // n*512 + k
    if (idx < H_ * D_) {
        int n = idx >> 9;
        int k = idx & 511;
        W1T[idx] = (__bf16)W1[k * H_ + n];
        W2T[idx] = (__bf16)W2[k * H_ + n];
    }
}

// ---------------------------------------------------------------------------
// Kernel B: h = relu(x @ W1 + b1), written bf16 to workspace.
// 256 threads = 8 waves; WG tile M=128 x N=64; each wave: 16(M) x 64(N),
// K streamed in 32-chunks. grid.x = N-tiles (fast) so the 8 WGs sharing an
// M-tile hit the same x rows back-to-back -> x served once from HBM via L2.
// A fragment (16-bit A 16x32, ISA 7.12.2): lane L holds row L%16,
//   K = k0 + (L/16)*8 + {0..7} and k0 + 16 + (L/16)*8 + {0..7}.
// B fragment (16-bit B 32x16): lane L holds col L%16,
//   K = k0 + (L/16)*8 + {0..7} and k0 + 16 + (L/16)*8 + {0..7} (as A; 2x8 contiguous in W1T[n][k]).
// ---------------------------------------------------------------------------
__global__ __launch_bounds__(256, 1) void gemm1_kernel(
    const float* __restrict__ x, const float* __restrict__ b1,
    const __bf16* __restrict__ W1T, __bf16* __restrict__ h) {
    const int tid  = threadIdx.x;
    const int wave = tid >> 5;
    const int lane = tid & 31;
    const int ln16 = lane & 15;
    const int lhi  = lane >> 4;
    const int m0   = blockIdx.y * 128 + wave * 16;
    const int c0   = blockIdx.x * 64;

    v8f acc[4];
    for (int j = 0; j < 4; ++j)
        for (int v = 0; v < 8; ++v) acc[j][v] = 0.0f;

    const float* xr = x + (size_t)(m0 + ln16) * D_;

    for (int k0 = 0; k0 < D_; k0 += 32) {
        // A: load 16 fp32 of this row, convert to bf16 fragment
        float4 f0 = *(const float4*)(xr + k0 + lhi * 8);
        float4 f1 = *(const float4*)(xr + k0 + lhi * 8 + 4);
        float4 f2 = *(const float4*)(xr + k0 + 16 + lhi * 8);
        float4 f3 = *(const float4*)(xr + k0 + 16 + lhi * 8 + 4);
        union { v16bf v; __bf16 e[16]; } ua;
        ua.e[0]  = (__bf16)f0.x; ua.e[1]  = (__bf16)f0.y; ua.e[2]  = (__bf16)f0.z; ua.e[3]  = (__bf16)f0.w;
        ua.e[4]  = (__bf16)f1.x; ua.e[5]  = (__bf16)f1.y; ua.e[6]  = (__bf16)f1.z; ua.e[7]  = (__bf16)f1.w;
        ua.e[8]  = (__bf16)f2.x; ua.e[9]  = (__bf16)f2.y; ua.e[10] = (__bf16)f2.z; ua.e[11] = (__bf16)f2.w;
        ua.e[12] = (__bf16)f3.x; ua.e[13] = (__bf16)f3.y; ua.e[14] = (__bf16)f3.z; ua.e[15] = (__bf16)f3.w;

        for (int j = 0; j < 4; ++j) {
            const __bf16* bp = W1T + (size_t)(c0 + 16 * j + ln16) * D_ + k0 + lhi * 8;
            union { v16bf v; uint4 q[2]; } ub;
            ub.q[0] = *(const uint4*)(bp);
            ub.q[1] = *(const uint4*)(bp + 16);
            acc[j] = __builtin_amdgcn_wmma_f32_16x16x32_bf16(
                false, ua.v, false, ub.v, (short)0, acc[j], false, false);
        }
    }

    // epilogue: + b1, relu, store bf16. C/D layout: lane L, VGPR v ->
    // row = 8*(L/16)+v, col = L%16 (cols contiguous across lanes -> 32B bursts)
    for (int j = 0; j < 4; ++j) {
        const int c  = c0 + 16 * j + ln16;
        const float bv = b1[c];
        for (int v = 0; v < 8; ++v) {
            const int m = m0 + lhi * 8 + v;
            float yv = fmaxf(acc[j][v] + bv, 0.0f);
            h[(size_t)m * H_ + c] = (__bf16)yv;
        }
    }
}

#if USE_TDM
// Issue a TDM load of one h tile: 2-D, tile 512(cols) x 16(rows), bf16,
// row stride T*H elements, LDS padded 4 dwords per 256 dwords (-> STRIDE=520).
// This toolchain exposes the 6-operand builtin
//   (uint32x4 g0, int32x8 g1, int32x4 g2, int32x4 g3, int32x8 g4, i32 cpol);
// our descriptor is fully contained in groups 0-1 (2-D tile), so the
// remaining operands are all-zero ("unused" per the ISA D# rules).
__device__ __forceinline__ void tdm_load_tile(const __bf16* gsrc, unsigned lds_off) {
    unsigned long long ga = (unsigned long long)(uintptr_t)gsrc;
    v4u g0 = { 1u,                                   // count=1 (valid user D#)
               lds_off,                              // lds_addr (bytes)
               (unsigned)ga,                         // global_addr[31:0]
               ((unsigned)((ga >> 32) & 0x01FFFFFFu)) | (2u << 30) }; // addr[56:32] | type=2
    v8i_t g1 = { (int)0x07D10000,      // data_size=2B | pad_en | pad_interval=256dw | pad_amount=4dw
                 (int)(512u << 16),    // tensor_dim0 = 512
                 (int)(16u << 16),     // tensor_dim1 = 16
                 (int)(512u << 16),    // tile_dim0   = 512
                 16,                   // tile_dim1=16, tile_dim2=0
                 (int)(T_ * H_),       // tensor_dim0_stride = 524288 (row -> row)
                 0, 0 };               // dim1_stride unused (2-D tile)
    v4i_t g2 = { 0, 0, 0, 0 };
    v4i_t g3 = { 0, 0, 0, 0 };
    v8i_t g4 = { 0, 0, 0, 0, 0, 0, 0, 0 };
    __builtin_amdgcn_tensor_load_to_lds(g0, g1, g2, g3, g4, 0);
}
#endif

// ---------------------------------------------------------------------------
// Kernel C: persistent scan. 8 WGs x 256 threads; WG g owns batch rows
// [16g, 16g+16). Per step t:
//   y_t = h_t + relu(y_{t-1} @ W2 + b2)   (wave w computes cols 64w..64w+63)
//   out[b,t] = sigmoid(y_t . Wc + bc)
// y state double-buffered bf16 in LDS (A operand read straight as fragments);
// h tile for step t+1 TDM-prefetched into LDS one step ahead.
// ---------------------------------------------------------------------------
__global__ __launch_bounds__(256, 1) void rnn_scan_kernel(
    const float* __restrict__ b2, const float* __restrict__ Wc,
    const float* __restrict__ bc, const __bf16* __restrict__ W2T,
    const __bf16* __restrict__ hsrc, float* __restrict__ out) {
    __shared__ __align__(16) __bf16 ybuf[2][16 * STRIDE];
    __shared__ __align__(16) __bf16 hbuf[2][16 * STRIDE];

    const int tid  = threadIdx.x;
    const int wave = tid >> 5;
    const int lane = tid & 31;
    const int ln16 = lane & 15;
    const int lhi  = lane >> 4;
    const int b0   = blockIdx.x * 16;
    const int c0   = wave * 64;

    // y_0 = 0
    for (int i = tid; i < 2 * 16 * STRIDE; i += 256) ybuf[0][i] = (__bf16)0.0f;

    float b2v[4];
    for (int j = 0; j < 4; ++j) b2v[j] = b2[c0 + 16 * j + ln16];

#if USE_TDM
    const unsigned hoff0 = (unsigned)(uintptr_t)&hbuf[0][0];
    const unsigned hoff1 = (unsigned)(uintptr_t)&hbuf[1][0];
    if (tid == 0) {
        tdm_load_tile(hsrc + (size_t)b0 * T_ * H_, hoff0);   // tile for t=0
        __builtin_amdgcn_s_wait_tensorcnt(0);
    }
#endif
    __syncthreads();

    for (int t = 0; t < T_; ++t) {
        const int rb = t & 1;        // y_{t-1} buffer / h_t buffer
        const int wb = rb ^ 1;       // y_t buffer

#if USE_TDM
        // prefetch h_{t+1}; hidden behind this whole step, waited before 6b
        if (tid == 0 && (t + 1) < T_)
            tdm_load_tile(hsrc + ((size_t)b0 * T_ + (t + 1)) * H_,
                          ((t + 1) & 1) ? hoff1 : hoff0);
#endif

        v8f acc[4];
        for (int j = 0; j < 4; ++j)
            for (int v = 0; v < 8; ++v) acc[j][v] = b2v[j];   // bias pre-loaded

        // GEMM: acc += y_{t-1} @ W2  (A from LDS, B streamed from L2-resident W2T)
        const __bf16* yr = &ybuf[rb][ln16 * STRIDE];
        for (int k0 = 0; k0 < H_; k0 += 32) {
            union { v16bf v; uint4 q[2]; } ua;
            ua.q[0] = *(const uint4*)(yr + k0 + lhi * 8);
            ua.q[1] = *(const uint4*)(yr + k0 + 16 + lhi * 8);
            for (int j = 0; j < 4; ++j) {
                const __bf16* bp = W2T + (size_t)(c0 + 16 * j + ln16) * H_ + k0 + lhi * 8;
                union { v16bf v; uint4 q[2]; } ub;
                ub.q[0] = *(const uint4*)(bp);
                ub.q[1] = *(const uint4*)(bp + 16);
                acc[j] = __builtin_amdgcn_wmma_f32_16x16x32_bf16(
                    false, ua.v, false, ub.v, (short)0, acc[j], false, false);
            }
        }

        // y_t = h_t + relu(acc); write bf16 into ybuf[wb]
        for (int j = 0; j < 4; ++j) {
            const int c = c0 + 16 * j + ln16;
            for (int v = 0; v < 8; ++v) {
                const int r = lhi * 8 + v;
#if USE_TDM
                float hv = (float)hbuf[rb][r * STRIDE + c];
#else
                float hv = (float)hsrc[((size_t)(b0 + r) * T_ + t) * H_ + c];
#endif
                float yv = hv + fmaxf(acc[j][v], 0.0f);
                ybuf[wb][r * STRIDE + c] = (__bf16)yv;
            }
        }
        __syncthreads();   // y_t complete & visible

        // classifier: 16 threads per row; partials overlaid on dead y_{t-1}
        float* dots = (float*)&ybuf[rb][0];
        {
            const int r = tid >> 4, sub = tid & 15;
            float p = 0.0f;
            for (int q = 0; q < 32; ++q) {
                const int k = sub + (q << 4);
                p += (float)ybuf[wb][r * STRIDE + k] * Wc[k];
            }
            dots[r * 16 + sub] = p;
        }
        __syncthreads();
        if (tid < 16) {
            float s = bc[0];
            for (int q = 0; q < 16; ++q) s += dots[tid * 16 + q];
            out[(size_t)(b0 + tid) * T_ + t] = 1.0f / (1.0f + __expf(-s));
        }
#if USE_TDM
        if (tid == 0) __builtin_amdgcn_s_wait_tensorcnt(0);  // h_{t+1} landed
#endif
        __syncthreads();   // protects dots region + publishes hbuf for next step
    }
}

// ---------------------------------------------------------------------------
// Launch. Workspace layout: [0,512K) W1T bf16 | [512K,1M) W2T bf16 |
// [1M, 1M+128M) h bf16 (B*T*H = 134,217,728 bytes).
// ---------------------------------------------------------------------------
extern "C" void kernel_launch(void* const* d_in, const int* in_sizes, int n_in,
                              void* d_out, int out_size, void* d_ws, size_t ws_size,
                              hipStream_t stream) {
    const float* x  = (const float*)d_in[0];
    const float* W1 = (const float*)d_in[1];
    const float* b1 = (const float*)d_in[2];
    const float* W2 = (const float*)d_in[3];
    const float* b2 = (const float*)d_in[4];
    const float* Wc = (const float*)d_in[5];
    const float* bc = (const float*)d_in[6];
    float* out = (float*)d_out;

    char* ws = (char*)d_ws;
    __bf16* W1T = (__bf16*)(ws);
    __bf16* W2T = (__bf16*)(ws + 512 * 1024);
    __bf16* h   = (__bf16*)(ws + 1024 * 1024);

    prep_kernel<<<(H_ * D_ + 255) / 256, 256, 0, stream>>>(W1, W2, W1T, W2T);

    dim3 gB(H_ / 64, (B_ * T_) / 128, 1);   // (8, 1024)
    gemm1_kernel<<<gB, 256, 0, stream>>>(x, b1, W1T, h);

    rnn_scan_kernel<<<B_ / 16, 256, 0, stream>>>(b2, Wc, bc, W2T, h, out);
}
